// Encoder_9663676416840
// MI455X (gfx1250) — matrix operand order repaired and checked by reading, not compiled, not run on hardware
//
#include <hip/hip_runtime.h>
#include <math.h>

#define NN     8192
#define NFEAT  512
#define NHID   256
#define KK     64
#define CAP    128

typedef __attribute__((ext_vector_type(16))) _Float16 v16h;
typedef __attribute__((ext_vector_type(8)))  float    v8f;

union HFrag { v16h v; _Float16 e[16]; };
union FFrag { v8f  v; float    e[8];  };

__device__ inline v8f wmma_f16(v16h a, v16h b, v8f c) {
  // D = A(16x32 f16) * B(32x16 f16) + C(16x16 f32)
  return __builtin_amdgcn_wmma_f32_16x16x32_f16(false, a, false, b, (short)0, c,
                                                false, false);
}

// CDNA5 async global->LDS copy (16B per lane), tracked by ASYNCcnt.
// vdst = LDS byte address (low 32 bits of flat LDS pointer), vaddr = global addr.
__device__ inline void async_ld_b128(unsigned lds_addr, const void* gaddr) {
  asm volatile("global_load_async_to_lds_b128 %0, %1, off"
               :: "v"(lds_addr), "v"(gaddr) : "memory");
}
__device__ inline void wait_async0() {
  asm volatile("s_wait_asynccnt 0x0" ::: "memory");
}

// ---------------- graph preprocessing ----------------

__global__ void k_init(float* deg, int* cnt, float* rowsum) {
  int i = blockIdx.x * blockDim.x + threadIdx.x;
  if (i < NN) { deg[i] = 1.0f; cnt[i] = 0; rowsum[i] = 0.0f; }  // deg=1 folds self-loop
}

__global__ void k_deg(const long long* col, int E, float* deg) {
  int i = blockIdx.x * blockDim.x + threadIdx.x;
  if (i < E) atomicAdd(&deg[(int)col[i]], 1.0f);
}

__global__ void k_dis(const float* deg, float* dis) {
  int i = blockIdx.x * blockDim.x + threadIdx.x;
  if (i < NN) dis[i] = rsqrtf(deg[i]);   // deg >= 1 always
}

__global__ void k_cand(const long long* row, const long long* col, int E,
                       const float* dis, float* rowsum, int* cnt,
                       int* cidx, float* cw) {
  int i = blockIdx.x * blockDim.x + threadIdx.x;
  if (i >= E + NN) return;
  int r, c;
  if (i < E) { r = (int)row[i]; c = (int)col[i]; }
  else       { r = i - E; c = r; }                 // self loop
  float w = dis[r] * dis[c];
  atomicAdd(&rowsum[r], w);
  int p = atomicAdd(&cnt[r], 1);
  if (p < CAP) { cidx[r * CAP + p] = c; cw[r * CAP + p] = w; }
}

// rank-select top-K per node (block of CAP threads per node)
__global__ __launch_bounds__(CAP) void k_topk(const int* cnt, const int* cidx,
                                              const float* cw, int* tki, float* tkw) {
  __shared__ float wl[CAP];
  __shared__ int   il[CAP];
  int n = blockIdx.x, t = threadIdx.x;
  int C = cnt[n]; if (C > CAP) C = CAP;
  if (t < C) { wl[t] = cw[n * CAP + t]; il[t] = cidx[n * CAP + t]; }
  if (t < KK) { tkw[n * KK + t] = 0.0f; tki[n * KK + t] = n; }   // zero-weight padding
  __syncthreads();
  if (t < C) {
    float wt = wl[t];
    int rank = 0;
    for (int j = 0; j < C; j++) {
      float wj = wl[j];
      rank += (wj > wt) || (wj == wt && j < t);
    }
    if (rank < KK) { tkw[n * KK + rank] = wt; tki[n * KK + rank] = il[t]; }
  }
}

// ---------------- WMMA GEMM: C[M x Nc] = A[M x Kd] * B[Kd x Nc] ----------------
// 64x64 block tile, 8 waves x two 16x16 WMMA tiles. A/B f32 tiles staged in LDS
// via async global->LDS b128 transfers, converted f32->f16 during fragment build.

#define BM 64
#define BN 64

__global__ __launch_bounds__(256) void k_gemm(const float* __restrict__ A,
                                              const float* __restrict__ B,
                                              float* __restrict__ C,
                                              int M, int Kd, int Nc) {
  __shared__ __align__(16) float Af[BM][36];      // 64x32 + pad (144B row stride)
  __shared__ __align__(16) float Bf[32][BN + 4];  // 32x64 + pad (272B row stride)

  int t = threadIdx.x;
  int wave = t >> 5, lane = t & 31;
  int m = lane & 15, hi = lane >> 4;

  int nBlkN = Nc / BN;
  int m0 = (blockIdx.x / nBlkN) * BM;
  int n0 = (blockIdx.x % nBlkN) * BN;

  // wave owns tiles `wave` and `wave+8` of the 4x4 grid of 16x16 tiles
  int ti0 = wave >> 2, tj0 = wave & 3;
  int ti1 = (wave + 8) >> 2, tj1 = (wave + 8) & 3;

  FFrag acc0, acc1;
#pragma unroll
  for (int v = 0; v < 8; v++) { acc0.e[v] = 0.0f; acc1.e[v] = 0.0f; }

  for (int k0 = 0; k0 < Kd; k0 += 32) {
    // ---- async stage: A 64x32 f32 (512 x 16B) + B 32x64 f32 (512 x 16B) ----
#pragma unroll
    for (int rep = 0; rep < 2; rep++) {
      int chunk = t + rep * 256;             // 0..511
      int ra = chunk >> 3, ca = (chunk & 7) << 2;     // A: row 0..63, col 0/4/../28
      async_ld_b128((unsigned)(uintptr_t)&Af[ra][ca],
                    A + (size_t)(m0 + ra) * Kd + k0 + ca);
      int rb = chunk >> 4, cb = (chunk & 15) << 2;    // B: row 0..31, col 0/4/../60
      async_ld_b128((unsigned)(uintptr_t)&Bf[rb][cb],
                    B + (size_t)(k0 + rb) * Nc + n0 + cb);
    }
    wait_async0();
    __syncthreads();

    // ---- fragments + WMMA ----
    HFrag a0, a1, b0, b1;
#pragma unroll
    for (int i = 0; i < 16; i++) {
      int ka = (i < 8) ? (hi * 8 + i) : (16 + hi * 8 + (i - 8));
      a0.e[i] = (_Float16)Af[ti0 * 16 + m][ka];
      a1.e[i] = (_Float16)Af[ti1 * 16 + m][ka];
      int kb = (i < 8) ? (hi * 8 + i) : (16 + hi * 8 + (i - 8));
      b0.e[i] = (_Float16)Bf[kb][tj0 * 16 + m];
      b1.e[i] = (_Float16)Bf[kb][tj1 * 16 + m];
    }
    acc0.v = wmma_f16(a0.v, b0.v, acc0.v);
    acc1.v = wmma_f16(a1.v, b1.v, acc1.v);
    __syncthreads();   // protect LDS before next async stage
  }

#pragma unroll
  for (int v = 0; v < 8; v++) {
    C[(size_t)(m0 + ti0 * 16 + v + hi * 8) * Nc + n0 + tj0 * 16 + m] = acc0.e[v];
    C[(size_t)(m0 + ti1 * 16 + v + hi * 8) * Nc + n0 + tj1 * 16 + m] = acc1.e[v];
  }
}

// ---------------- soft-k-medoid layer (one 256-thread WG per node) ----------------

__global__ __launch_bounds__(256) void k_medoid(const float* __restrict__ X,
                                                const int* __restrict__ tki,
                                                const float* __restrict__ tkw,
                                                const float* __restrict__ rowsum,
                                                const float* __restrict__ bias,
                                                float* __restrict__ out) {
  __shared__ __align__(16) _Float16 xk[KK][NHID + 8];  // 64x256 f16, 528B row stride
  __shared__ float G[KK][KK + 1];                      // Gram (diag = squared norms)
  __shared__ float aw[KK];
  __shared__ int   ids[KK];
  __shared__ float cv[KK];
  __shared__ float sw[KK];
  __shared__ float s_min, s_sum;

  int n = blockIdx.x, t = threadIdx.x;
  if (t < KK) { aw[t] = tkw[n * KK + t]; ids[t] = tki[n * KK + t]; }
  __syncthreads();

  // gather top-K neighbor rows (float4 loads), convert to f16
  for (int i = t; i < KK * (NHID / 4); i += 256) {
    int k  = i >> 6;              // NHID/4 = 64 chunks per row
    int d4 = (i & 63) << 2;
    const float4 f = *(const float4*)(X + (size_t)ids[k] * NHID + d4);
    xk[k][d4 + 0] = (_Float16)f.x;
    xk[k][d4 + 1] = (_Float16)f.y;
    xk[k][d4 + 2] = (_Float16)f.z;
    xk[k][d4 + 3] = (_Float16)f.w;
  }
  __syncthreads();

  // Gram = xk (64x256) * xk^T via 16 WMMA tiles, 2 per wave
  int wave = t >> 5, lane = t & 31;
  int m = lane & 15, hi = lane >> 4;
  for (int tile = wave; tile < 16; tile += 8) {
    int ti = tile >> 2, tj = tile & 3;
    FFrag c;
#pragma unroll
    for (int v = 0; v < 8; v++) c.e[v] = 0.0f;
    for (int k0 = 0; k0 < NHID; k0 += 32) {
      HFrag a, b;
#pragma unroll
      for (int i = 0; i < 16; i++) {
        int ka = (i < 8) ? (hi * 8 + i) : (16 + hi * 8 + (i - 8));
        a.e[i] = xk[ti * 16 + m][k0 + ka];
        int kb = (i < 8) ? (hi * 8 + i) : (16 + hi * 8 + (i - 8));  // B[kk][nn] = xk[tj*16+nn][k0+kk]
        b.e[i] = xk[tj * 16 + m][k0 + kb];
      }
      c.v = wmma_f16(a.v, b.v, c.v);
    }
#pragma unroll
    for (int v = 0; v < 8; v++)
      G[ti * 16 + v + hi * 8][tj * 16 + m] = c.e[v];
  }
  __syncthreads();

  // c[j] = sum_l aw[l] * ||x_l - x_j||  (sq norms = Gram diagonal)
  if (t < KK) {
    float sqj = G[t][t];
    float cacc = 0.0f;
    for (int l = 0; l < KK; l++) {
      float d2 = G[l][l] + sqj - 2.0f * G[l][t];
      cacc += aw[l] * sqrtf(fmaxf(d2, 0.0f));
    }
    cv[t] = (aw[t] == 0.0f) ? 3.0e38f : cacc;
  }
  __syncthreads();

  if (t == 0) {
    float mn = 3.0e38f;
    for (int j = 0; j < KK; j++)
      if (aw[j] > 0.0f && cv[j] < mn) mn = cv[j];
    s_min = mn;
  }
  __syncthreads();
  // softmax(-c) * aw   (TEMP = 1, with_weight_correction)
  if (t < KK) sw[t] = (aw[t] > 0.0f) ? expf(-(cv[t] - s_min)) * aw[t] : 0.0f;
  __syncthreads();
  if (t == 0) {
    float s = 0.0f;
    for (int j = 0; j < KK; j++) s += sw[j];
    s_sum = s;
  }
  __syncthreads();

  // out[n][t] = relu(rowsum * sum_k soft[k]*xk[k][t] + bias[t])
  float acc = 0.0f;
  for (int k = 0; k < KK; k++) acc += sw[k] * (float)xk[k][t];
  float val = rowsum[n] * (acc / s_sum) + bias[t];
  out[(size_t)n * NHID + t] = fmaxf(val, 0.0f);
}

// ---------------- host launcher ----------------

extern "C" void kernel_launch(void* const* d_in, const int* in_sizes, int n_in,
                              void* d_out, int out_size, void* d_ws, size_t ws_size,
                              hipStream_t stream) {
  const float*     x   = (const float*)d_in[0];
  const long long* ei  = (const long long*)d_in[1];
  const float*     W1  = (const float*)d_in[2];
  const float*     b1  = (const float*)d_in[3];
  const float*     W2  = (const float*)d_in[4];
  const float*     b2  = (const float*)d_in[5];
  float*           out = (float*)d_out;

  int E = in_sizes[1] / 2;
  const long long* erow = ei;
  const long long* ecol = ei + E;

  char* w = (char*)d_ws;
  size_t off = 0;
  auto alloc = [&](size_t bytes) -> void* {
    void* p = w + off;
    off = (off + bytes + 255) & ~(size_t)255;
    return p;
  };
  float* deg    = (float*)alloc((size_t)NN * 4);
  float* dis    = (float*)alloc((size_t)NN * 4);
  float* rowsum = (float*)alloc((size_t)NN * 4);
  int*   cnt    = (int*)  alloc((size_t)NN * 4);
  int*   cidx   = (int*)  alloc((size_t)NN * CAP * 4);
  float* cw     = (float*)alloc((size_t)NN * CAP * 4);
  int*   tki    = (int*)  alloc((size_t)NN * KK * 4);
  float* tkw    = (float*)alloc((size_t)NN * KK * 4);
  float* buf1   = (float*)alloc((size_t)NN * NHID * 4);
  float* buf2   = (float*)alloc((size_t)NN * NHID * 4);

  // graph normalization + top-K (shared by both layers)
  k_init<<<NN / 256, 256, 0, stream>>>(deg, cnt, rowsum);
  k_deg<<<(E + 255) / 256, 256, 0, stream>>>(ecol, E, deg);
  k_dis<<<NN / 256, 256, 0, stream>>>(deg, dis);
  k_cand<<<(E + NN + 255) / 256, 256, 0, stream>>>(erow, ecol, E, dis, rowsum, cnt, cidx, cw);
  k_topk<<<NN, CAP, 0, stream>>>(cnt, cidx, cw, tki, tkw);

  // layer 1: x @ W1 -> medoid -> +b1, relu
  int blocks1 = (NN / BM) * (NHID / BN);
  k_gemm<<<blocks1, 256, 0, stream>>>(x, W1, buf1, NN, NFEAT, NHID);
  k_medoid<<<NN, 256, 0, stream>>>(buf1, tki, tkw, rowsum, b1, buf2);

  // layer 2: h @ W2 -> medoid -> +b2, relu
  k_gemm<<<blocks1, 256, 0, stream>>>(buf2, W2, buf1, NN, NHID, NHID);
  k_medoid<<<NN, 256, 0, stream>>>(buf1, tki, tkw, rowsum, b2, out);
}
